// GatedAttention_43928925504191
// MI455X (gfx1250) — matrix operand order repaired and checked by reading, not compiled, not run on hardware
//
#include <hip/hip_runtime.h>
#include <hip/hip_bf16.h>

// B=32, C=512, T=1024 gated channel-attention, bf16 WMMA pipeline:
//  K0: x [B,C,T] f32 -> xt [B,T,C] bf16 (transpose+convert)
//  K1: q/k = gate*(xt@W^T+b) stored transposed [B,C,T]; v stored [B,T,C]
//  K2: energy[b,c,d] = SCALE * sum_t q[t,c]k[t,d]  (f32 [B,C,C])
//  K3: row softmax over d -> attn bf16 [B,C,C]
//  K4: out[b,c,t] = sum_d attn[c,d] v[t,d]  (f32 [B,C,T])
// GEMMs: 256(m)x64(n) block tile, 8 waves x (32x64) each, K slab 32,
// double-buffered LDS staged with global_load_async_to_lds_b128 (ASYNCcnt).

#define BB 32
#define CC 512
#define TT 1024
#define SCALE 0.044194173824159216f  // 1/sqrt(512)
#define LDA 40                       // LDS row stride in bf16 elems (80B)

typedef __attribute__((ext_vector_type(16))) __bf16 v16bf;
typedef __attribute__((ext_vector_type(8)))  float  v8f;
typedef int v4i_ga __attribute__((vector_size(16)));

union v16bf_u { v16bf v; uint4 q[2]; };
union pk8_u   { __bf16 h[8]; uint4 u; };
union pk4_u   { __bf16 h[4]; uint2 u; };

// ---- CDNA5 async global->LDS staging (ASYNCcnt path), guarded --------------
#if __has_builtin(__builtin_amdgcn_global_load_async_to_lds_b128) && \
    __has_builtin(__builtin_amdgcn_s_wait_asynccnt)
#define GA_ASYNC 1
typedef __attribute__((address_space(1))) v4i_ga* ga_gp_t;  // global int4*
typedef __attribute__((address_space(3))) v4i_ga* ga_lp_t;  // LDS int4*
#endif

__device__ __forceinline__ void copy16_g2l(const __bf16* g, __bf16* l) {
#ifdef GA_ASYNC
  __builtin_amdgcn_global_load_async_to_lds_b128((ga_gp_t)g, (ga_lp_t)l, 0, 0);
#else
  *(uint4*)l = *(const uint4*)g;
#endif
}

__device__ __forceinline__ void stage_fence() {
#ifdef GA_ASYNC
  __builtin_amdgcn_s_wait_asynccnt(0);
#endif
  __syncthreads();
}

// A fragment (16x32 MxK) from row-major [m][k] LDS tile:
// lane(m,h): e0..7 <- k=8h..8h+7 ; e8..15 <- k=16+8h..16+8h+7  => 2x b128
__device__ __forceinline__ v16bf load_frag_a(const __bf16* lds, int m_base) {
  const int lane = threadIdx.x & 31;
  const int h = lane >> 4;
  const __bf16* row = lds + (size_t)((lane & 15) + m_base) * LDA;
  v16bf_u u;
  u.q[0] = *(const uint4*)(row + 8 * h);
  u.q[1] = *(const uint4*)(row + 16 + 8 * h);
  return u.v;
}

// B fragment (32x16 KxN) from row-major [n][k] LDS tile:
// lane(n,h): e0..7 <- k=8h..8h+7 ; e8..15 <- k=16+8h..16+8h+7  => 2x b128
__device__ __forceinline__ v16bf load_frag_b(const __bf16* lds, int n_base) {
  const int lane = threadIdx.x & 31;
  const int h = lane >> 4;
  const __bf16* row = lds + (size_t)((lane & 15) + n_base) * LDA;
  v16bf_u u;
  u.q[0] = *(const uint4*)(row + 8 * h);
  u.q[1] = *(const uint4*)(row + 16 + 8 * h);
  return u.v;
}

__device__ __forceinline__ v8f wmma_bf16(v16bf a, v16bf b, v8f c) {
  return __builtin_amdgcn_wmma_f32_16x16x32_bf16(false, a, false, b, (short)0, c,
                                                 false, false);
}

// ---- K0: transpose + convert: x [B,C,T] f32 -> xt [B,T,C] bf16 -------------
__global__ __launch_bounds__(256) void ga_transpose_kernel(
    const float* __restrict__ x, __bf16* __restrict__ xt) {
  __shared__ __bf16 tile[32][36];
  const int b  = blockIdx.y;
  const int t0 = (blockIdx.x & 31) * 32;
  const int c0 = (blockIdx.x >> 5) * 32;
  const int tid = threadIdx.x;
  {
    const int c = tid >> 3, tq = (tid & 7) * 4;
    float4 f = *(const float4*)(x + ((size_t)b * CC + c0 + c) * TT + t0 + tq);
    tile[c][tq + 0] = (__bf16)f.x;
    tile[c][tq + 1] = (__bf16)f.y;
    tile[c][tq + 2] = (__bf16)f.z;
    tile[c][tq + 3] = (__bf16)f.w;
  }
  __syncthreads();
  {
    const int t = tid >> 3, cq = (tid & 7) * 4;
    pk4_u pk;
#pragma unroll
    for (int i = 0; i < 4; ++i) pk.h[i] = tile[cq + i][t];
    *(uint2*)(xt + ((size_t)b * TT + t0 + t) * CC + c0 + cq) = pk.u;
  }
}

// ---- K1: q/k/v projection. Tile 256(t) x 64(d), wave = 32x64, K slab 32 ----
__global__ __launch_bounds__(256) void ga_qkv_kernel(
    const __bf16* __restrict__ xt,
    const float* __restrict__ gq, const float* __restrict__ gk,
    const float* __restrict__ gv,
    const float* __restrict__ Wq, const float* __restrict__ bq,
    const float* __restrict__ Wk, const float* __restrict__ bk,
    const float* __restrict__ Wv, const float* __restrict__ bv,
    __bf16* __restrict__ qT, __bf16* __restrict__ kT,
    __bf16* __restrict__ vbuf) {
  const int b  = blockIdx.y;
  const int p  = blockIdx.z;
  const int t0 = (blockIdx.x & 3) * 256;
  const int d0 = (blockIdx.x >> 2) * 64;

  const float* W; const float* bias; const float* gate; __bf16* outp;
  if (p == 0)      { W = Wq; bias = bq; gate = gq; outp = qT; }
  else if (p == 1) { W = Wk; bias = bk; gate = gk; outp = kT; }
  else             { W = Wv; bias = bv; gate = gv; outp = vbuf; }

  __shared__ __bf16 lds_a[2][256][LDA];  // [m=t][k=c], double-buffered
  __shared__ __bf16 lds_b[2][64][LDA];   // [n=d][k=c]

  const int tid = threadIdx.x;
  const int wave = tid >> 5;

  auto stage = [&](int buf, int kc) {
    {  // A: one 64B row per thread, async copy
      const __bf16* srow = xt + ((size_t)b * TT + t0 + tid) * CC + kc;
      __bf16* drow = &lds_a[buf][tid][0];
#pragma unroll
      for (int i = 0; i < 4; ++i) copy16_g2l(srow + 8 * i, drow + 8 * i);
    }
    {  // B: W rows f32 -> bf16, packed single b128 store
      const int d = tid >> 2, kq = (tid & 3) * 8;
      const float4* src = (const float4*)(W + (size_t)(d0 + d) * CC + kc + kq);
      float4 f0 = src[0], f1 = src[1];
      pk8_u pk;
      pk.h[0] = (__bf16)f0.x; pk.h[1] = (__bf16)f0.y;
      pk.h[2] = (__bf16)f0.z; pk.h[3] = (__bf16)f0.w;
      pk.h[4] = (__bf16)f1.x; pk.h[5] = (__bf16)f1.y;
      pk.h[6] = (__bf16)f1.z; pk.h[7] = (__bf16)f1.w;
      *(uint4*)&lds_b[buf][d][kq] = pk.u;
    }
  };

  v8f acc[2][4];
#pragma unroll
  for (int ms = 0; ms < 2; ++ms)
#pragma unroll
    for (int s = 0; s < 4; ++s) acc[ms][s] = (v8f){0.f,0.f,0.f,0.f,0.f,0.f,0.f,0.f};

  stage(0, 0);
  stage_fence();
  int cur = 0;
  for (int kc = 0; kc < CC; kc += 32) {
    if (kc + 32 < CC) stage(cur ^ 1, kc + 32);  // DMA next slab under compute
    v16bf af0 = load_frag_a(&lds_a[cur][0][0], wave * 32);
    v16bf af1 = load_frag_a(&lds_a[cur][0][0], wave * 32 + 16);
#pragma unroll
    for (int s = 0; s < 4; ++s) {
      v16bf bf = load_frag_b(&lds_b[cur][0][0], s * 16);
      acc[0][s] = wmma_bf16(af0, bf, acc[0][s]);
      acc[1][s] = wmma_bf16(af1, bf, acc[1][s]);
    }
    stage_fence();
    cur ^= 1;
  }

  // Epilogue. D-frag: reg r -> t = tbase + r (consecutive!), lane -> d.
  const int lane = tid & 31;
  const int h = lane >> 4;
  const int n = lane & 15;
#pragma unroll
  for (int ms = 0; ms < 2; ++ms) {
    const int tbase = t0 + wave * 32 + ms * 16 + 8 * h;
#pragma unroll
    for (int s = 0; s < 4; ++s) {
      const int gd = d0 + s * 16 + n;
      const float bs = bias[gd];
      const float4* gp = (const float4*)(gate + ((size_t)b * CC + gd) * TT + tbase);
      float4 g0 = gp[0], g1 = gp[1];
      float gr[8] = {g0.x, g0.y, g0.z, g0.w, g1.x, g1.y, g1.z, g1.w};
      if (p < 2) {
        // q/k transposed [B,C,T]: 8 consecutive t -> one b128 store
        pk8_u pk;
#pragma unroll
        for (int r = 0; r < 8; ++r) pk.h[r] = (__bf16)((acc[ms][s][r] + bs) * gr[r]);
        *(uint4*)(outp + ((size_t)b * CC + gd) * TT + tbase) = pk.u;
      } else {
        // v [B,T,C]: scatter over t
#pragma unroll
        for (int r = 0; r < 8; ++r) {
          float val = (acc[ms][s][r] + bs) * gr[r];
          outp[((size_t)b * TT + tbase + r) * CC + gd] = (__bf16)val;
        }
      }
    }
  }
}

// ---- K2: energy = SCALE * q^T k. Tile 256(c) x 64(d), K(=t) slab 32 --------
__global__ __launch_bounds__(256) void ga_energy_kernel(
    const __bf16* __restrict__ qT, const __bf16* __restrict__ kT,
    float* __restrict__ energy) {
  const int b  = blockIdx.y;
  const int c0 = (blockIdx.x & 1) * 256;
  const int d0 = (blockIdx.x >> 1) * 64;

  __shared__ __bf16 lds_a[2][256][LDA];  // [m=c][k=t]
  __shared__ __bf16 lds_b[2][64][LDA];   // [n=d][k=t]

  const int tid = threadIdx.x;
  const int wave = tid >> 5;

  auto stage = [&](int buf, int tc) {
    {  // A: qT row per thread (contiguous t)
      const __bf16* srow = qT + ((size_t)b * CC + c0 + tid) * TT + tc;
      __bf16* drow = &lds_a[buf][tid][0];
#pragma unroll
      for (int i = 0; i < 4; ++i) copy16_g2l(srow + 8 * i, drow + 8 * i);
    }
    {  // B: kT rows (contiguous t)
      const int nn = tid >> 2, kq = (tid & 3) * 8;
      copy16_g2l(kT + ((size_t)b * CC + d0 + nn) * TT + tc + kq,
                 &lds_b[buf][nn][kq]);
    }
  };

  v8f acc[2][4];
#pragma unroll
  for (int ms = 0; ms < 2; ++ms)
#pragma unroll
    for (int s = 0; s < 4; ++s) acc[ms][s] = (v8f){0.f,0.f,0.f,0.f,0.f,0.f,0.f,0.f};

  stage(0, 0);
  stage_fence();
  int cur = 0;
  for (int tc = 0; tc < TT; tc += 32) {
    if (tc + 32 < TT) stage(cur ^ 1, tc + 32);
    v16bf af0 = load_frag_a(&lds_a[cur][0][0], wave * 32);
    v16bf af1 = load_frag_a(&lds_a[cur][0][0], wave * 32 + 16);
#pragma unroll
    for (int s = 0; s < 4; ++s) {
      v16bf bf = load_frag_b(&lds_b[cur][0][0], s * 16);
      acc[0][s] = wmma_bf16(af0, bf, acc[0][s]);
      acc[1][s] = wmma_bf16(af1, bf, acc[1][s]);
    }
    stage_fence();
    cur ^= 1;
  }

  const int lane = tid & 31;
  const int h = lane >> 4;
  const int n = lane & 15;
#pragma unroll
  for (int ms = 0; ms < 2; ++ms)
#pragma unroll
    for (int s = 0; s < 4; ++s) {
      const int gd = d0 + s * 16 + n;
#pragma unroll
      for (int r = 0; r < 8; ++r) {
        const int gc = c0 + wave * 32 + ms * 16 + r + 8 * h;
        energy[((size_t)b * CC + gc) * CC + gd] = acc[ms][s][r] * SCALE;
      }
    }
}

// ---- K3: row softmax (rows of 512), bf16 out -------------------------------
__global__ __launch_bounds__(256) void ga_softmax_kernel(
    const float* __restrict__ energy, __bf16* __restrict__ attn) {
  const size_t row = blockIdx.x;  // b*C + c
  const float* e = energy + row * CC;
  __shared__ float red[256];
  const int tid = threadIdx.x;

  float v0 = e[tid], v1 = e[tid + 256];
  red[tid] = fmaxf(v0, v1);
  __syncthreads();
#pragma unroll
  for (int s = 128; s > 0; s >>= 1) {
    if (tid < s) red[tid] = fmaxf(red[tid], red[tid + s]);
    __syncthreads();
  }
  const float rmax = red[0];
  __syncthreads();

  const float e0 = __expf(v0 - rmax), e1 = __expf(v1 - rmax);
  red[tid] = e0 + e1;
  __syncthreads();
#pragma unroll
  for (int s = 128; s > 0; s >>= 1) {
    if (tid < s) red[tid] += red[tid + s];
    __syncthreads();
  }
  const float inv = 1.0f / red[0];
  attn[row * CC + tid]       = (__bf16)(e0 * inv);
  attn[row * CC + tid + 256] = (__bf16)(e1 * inv);
}

// ---- K4: out = attn @ v^T. Tile 256(c) x 64(t), K(=d) slab 32 --------------
__global__ __launch_bounds__(256) void ga_out_kernel(
    const __bf16* __restrict__ attn, const __bf16* __restrict__ vbuf,
    float* __restrict__ out) {
  const int b  = blockIdx.y;
  const int c0 = (blockIdx.x & 1) * 256;
  const int t0 = (blockIdx.x >> 1) * 64;

  __shared__ __bf16 lds_a[2][256][LDA];  // [m=c][k=d]
  __shared__ __bf16 lds_b[2][64][LDA];   // [n=t][k=d]

  const int tid = threadIdx.x;
  const int wave = tid >> 5;

  auto stage = [&](int buf, int dc) {
    {  // A: attn row per thread (contiguous d)
      const __bf16* srow = attn + ((size_t)b * CC + c0 + tid) * CC + dc;
      __bf16* drow = &lds_a[buf][tid][0];
#pragma unroll
      for (int i = 0; i < 4; ++i) copy16_g2l(srow + 8 * i, drow + 8 * i);
    }
    {  // B: v rows (contiguous d)
      const int nn = tid >> 2, kq = (tid & 3) * 8;
      copy16_g2l(vbuf + ((size_t)b * TT + t0 + nn) * CC + dc + kq,
                 &lds_b[buf][nn][kq]);
    }
  };

  v8f acc[2][4];
#pragma unroll
  for (int ms = 0; ms < 2; ++ms)
#pragma unroll
    for (int s = 0; s < 4; ++s) acc[ms][s] = (v8f){0.f,0.f,0.f,0.f,0.f,0.f,0.f,0.f};

  stage(0, 0);
  stage_fence();
  int cur = 0;
  for (int dc = 0; dc < CC; dc += 32) {
    if (dc + 32 < CC) stage(cur ^ 1, dc + 32);
    v16bf af0 = load_frag_a(&lds_a[cur][0][0], wave * 32);
    v16bf af1 = load_frag_a(&lds_a[cur][0][0], wave * 32 + 16);
#pragma unroll
    for (int s = 0; s < 4; ++s) {
      v16bf bf = load_frag_b(&lds_b[cur][0][0], s * 16);
      acc[0][s] = wmma_bf16(af0, bf, acc[0][s]);
      acc[1][s] = wmma_bf16(af1, bf, acc[1][s]);
    }
    stage_fence();
    cur ^= 1;
  }

  const int lane = tid & 31;
  const int h = lane >> 4;
  const int n = lane & 15;
#pragma unroll
  for (int ms = 0; ms < 2; ++ms)
#pragma unroll
    for (int s = 0; s < 4; ++s) {
      const int gt = t0 + s * 16 + n;
#pragma unroll
      for (int r = 0; r < 8; ++r) {
        const int gc = c0 + wave * 32 + ms * 16 + r + 8 * h;
        out[((size_t)b * CC + gc) * TT + gt] = acc[ms][s][r];
      }
    }
}

// ---- Launch ---------------------------------------------------------------
extern "C" void kernel_launch(void* const* d_in, const int* in_sizes, int n_in,
                              void* d_out, int out_size, void* d_ws,
                              size_t ws_size, hipStream_t stream) {
  (void)in_sizes; (void)n_in; (void)out_size; (void)ws_size;
  const float* x  = (const float*)d_in[0];
  const float* gq = (const float*)d_in[1];
  const float* gk = (const float*)d_in[2];
  const float* gv = (const float*)d_in[3];
  const float* Wq = (const float*)d_in[4];
  const float* bq = (const float*)d_in[5];
  const float* Wk = (const float*)d_in[6];
  const float* bk = (const float*)d_in[7];
  const float* Wv = (const float*)d_in[8];
  const float* bv = (const float*)d_in[9];
  float* out = (float*)d_out;

  char* ws = (char*)d_ws;
  const size_t SEG = (size_t)BB * TT * CC * sizeof(__bf16);  // 32 MB
  __bf16* xt     = (__bf16*)ws;              // dead after K1
  float*  energy = (float*)ws;               // overlays xt (B*C*C*4 = 32 MB)
  __bf16* qT     = (__bf16*)(ws + SEG);      // dead after K2
  __bf16* attn   = (__bf16*)(ws + SEG);      // overlays qT (16 MB)
  __bf16* kT     = (__bf16*)(ws + 2 * SEG);
  __bf16* vbuf   = (__bf16*)(ws + 3 * SEG);

  ga_transpose_kernel<<<dim3(512, BB), 256, 0, stream>>>(x, xt);
  ga_qkv_kernel<<<dim3(32, BB, 3), 256, 0, stream>>>(
      xt, gq, gk, gv, Wq, bq, Wk, bk, Wv, bv, qT, kT, vbuf);
  ga_energy_kernel<<<dim3(16, BB), 256, 0, stream>>>(qT, kT, energy);
  ga_softmax_kernel<<<dim3(BB * CC), 256, 0, stream>>>(energy, attn);
  ga_out_kernel<<<dim3(32, BB), 256, 0, stream>>>(attn, vbuf, out);
}
